// Self_Attn_23390391894736
// MI455X (gfx1250) — matrix operand order repaired and checked by reading, not compiled, not run on hardware
//
#include <hip/hip_runtime.h>
#include <hip/hip_bf16.h>

typedef __attribute__((ext_vector_type(16))) __bf16   v16bf;
typedef __attribute__((ext_vector_type(8)))  float    v8f;
typedef __attribute__((ext_vector_type(8)))  unsigned uv8;
typedef __attribute__((ext_vector_type(4)))  unsigned uv4;

#define BDIM 8
#define CDIM 256
#define NPIX 4096
#define CQKD 32
#define COD  256

// ---- TDM configuration (K tile staging) -----------------------------------
#define KPITCH 40                      // 32 elems + 4-DWORD pad -> 80B row pitch
#define KBYTES (64 * KPITCH * 2)       // 5120 B per K buffer
#define ATTN_SMEM (4 * KBYTES)         // 2 waves x 2 buffers

#ifndef __has_builtin
#define __has_builtin(x) 0
#endif
#if defined(__HIP_DEVICE_COMPILE__) && \
    __has_builtin(__builtin_amdgcn_tensor_load_to_lds) && \
    __has_builtin(__builtin_amdgcn_s_wait_tensorcnt)
#define USE_TDM 1
#if __has_include(<hip/amd_detail/amd_gfx1250_TDM.h>)
#define TDM6 1
#else
#define TDM6 0
#endif
#else
#define USE_TDM 0
#endif

// Native RNE f32->bf16 (lowers to v_cvt_pk_bf16_f32).
static __device__ __forceinline__ unsigned pack_bf16(float a, float b) {
    unsigned short ua = __builtin_bit_cast(unsigned short, (__bf16)a);
    unsigned short ub = __builtin_bit_cast(unsigned short, (__bf16)b);
    return (unsigned)ua | ((unsigned)ub << 16);
}
static __device__ __forceinline__ unsigned short bf16_bits(float a) {
    return __builtin_bit_cast(unsigned short, (__bf16)a);
}
// Concatenate two 16B bf16 chunks into one A/B fragment (pure b128 loads).
static __device__ __forceinline__ v16bf cat16(const unsigned short* p0,
                                              const unsigned short* p1) {
    uv4 lo = *(const uv4*)p0;
    uv4 hi = *(const uv4*)p1;
    uv8 c = {lo[0], lo[1], lo[2], lo[3], hi[0], hi[1], hi[2], hi[3]};
    return __builtin_bit_cast(v16bf, c);
}

#if USE_TDM
typedef __attribute__((ext_vector_type(4))) unsigned tg0_t;
typedef __attribute__((ext_vector_type(8))) int      tg1_t;
typedef __attribute__((ext_vector_type(4))) int      tg2_t;

// 2-D TDM tile load: tile_w elems/row (2B elems), tile_h rows, row stride in
// elems; LDS rows padded by pad_amt DWORDs every pad_int-coded DWORDs.
static __device__ __forceinline__ void tdm_load_2d(
    unsigned lds_off, const void* gaddr, unsigned tile_w, unsigned tile_h,
    unsigned row_stride, unsigned pad_int, unsigned pad_amt)
{
    const unsigned long long ga = (unsigned long long)gaddr;
    tg0_t g0;
    g0[0] = 1u;                                            // count=1, user mode
    g0[1] = lds_off;                                       // lds_addr
    g0[2] = (unsigned)(ga & 0xFFFFFFFFu);                  // global_addr lo
    g0[3] = (unsigned)((ga >> 32) & 0x01FFFFFFu) | 0x80000000u;  // addr hi | type=2

    const unsigned w0  = (1u << 16) | (1u << 20) | (pad_int << 22) | (pad_amt << 25);
    const unsigned td0 = 1u << 20;                         // huge: no OOB clipping
    const unsigned td1 = 1u << 20;
    tg1_t g1;
    g1[0] = (int)w0;                                       // mask=0,dsz=2B,pad_en
    g1[1] = (int)((td0 & 0xFFFFu) << 16);                  // tensor_dim0 lo16
    g1[2] = (int)((td0 >> 16) | ((td1 & 0xFFFFu) << 16));  // td0 hi16 | td1 lo16
    g1[3] = (int)((td1 >> 16) | (tile_w << 16));           // td1 hi16 | tile_dim0
    g1[4] = (int)(tile_h & 0xFFFFu);                       // tile_dim1, tile_dim2=0
    g1[5] = (int)row_stride;                               // tensor_dim0_stride lo32
    g1[6] = 0;
    g1[7] = 0;
    const tg2_t gz = {0, 0, 0, 0};
#if TDM6
    const tg1_t gz8 = {0, 0, 0, 0, 0, 0, 0, 0};
    __builtin_amdgcn_tensor_load_to_lds(g0, g1, gz, gz, gz8, 0);
#else
    __builtin_amdgcn_tensor_load_to_lds(g0, g1, gz, gz, 0);
#endif
}
#endif  // USE_TDM

// ---------------------------------------------------------------------------
// Kernel 1: fused Q/K/V projection (stacked 320x256 GEMM vs x).
// ---------------------------------------------------------------------------
__global__ __launch_bounds__(32) void proj_kernel(
    const float* __restrict__ x,
    const float* __restrict__ Wq, const float* __restrict__ bq,
    const float* __restrict__ Wk, const float* __restrict__ bk,
    const float* __restrict__ Wv, const float* __restrict__ bv,
    unsigned short* __restrict__ qws, unsigned short* __restrict__ kws,
    unsigned short* __restrict__ vws)
{
    const int lane = threadIdx.x & 31;
    const int hi   = lane >> 4;
    const int col  = lane & 15;
    const int n0   = blockIdx.x * 16;
    const int rt   = blockIdx.y;
    const int b    = blockIdx.z;

    const float* W; const float* bias; int rowbase;
    if (rt < 2)      { W = Wq; bias = bq; rowbase = rt * 16; }
    else if (rt < 4) { W = Wk; bias = bk; rowbase = (rt - 2) * 16; }
    else             { W = Wv; bias = bv; rowbase = (rt - 4) * 16; }

    const float* wrow = W + (size_t)(rowbase + col) * CDIM;
    v8f acc = {};
    for (int kc = 0; kc < CDIM / 32; ++kc) {
        const int c0 = kc * 32;
        unsigned au[8], bu[8];
        #pragma unroll
        for (int p = 0; p < 8; ++p) {
            const int ka = (p << 1) + (hi << 4);
            const float* wp = wrow + c0 + ka;
            au[p] = pack_bf16(wp[0], wp[1]);
            const int cb = (p << 1) + (hi << 4);
            const float* xp = x + ((size_t)b * CDIM + c0 + cb) * NPIX + n0 + col;
            bu[p] = pack_bf16(xp[0], xp[NPIX]);
        }
        v16bf a  = __builtin_bit_cast(v16bf, (uv8){au[0],au[1],au[2],au[3],au[4],au[5],au[6],au[7]});
        v16bf bb = __builtin_bit_cast(v16bf, (uv8){bu[0],bu[1],bu[2],bu[3],bu[4],bu[5],bu[6],bu[7]});
        acc = __builtin_amdgcn_wmma_f32_16x16x32_bf16(false, a, false, bb,
                                                      (short)0, acc, false, false);
    }
    #pragma unroll
    for (int v = 0; v < 8; ++v) {
        const int mrow = v + (hi << 3);
        const unsigned short h = bf16_bits(acc[v] + bias[rowbase + mrow]);
        const int gs = rt * 16 + mrow;
        if (gs < 32)
            qws[((size_t)b * NPIX + n0 + col) * CQKD + gs] = h;
        else if (gs < 64)
            kws[((size_t)b * NPIX + n0 + col) * CQKD + (gs - 32)] = h;
        else
            vws[((size_t)b * COD + (gs - 64)) * NPIX + n0 + col] = h;
    }
}

// ---------------------------------------------------------------------------
// Kernel 2: flash attention (S^T = K^T Q, O^T = V P^T), K tiles staged by the
// Tensor Data Mover into LDS with double buffering (TENSORcnt pipelining).
// Block = 64 threads = 2 waves; each wave: 16 queries x 128 output channels.
// ---------------------------------------------------------------------------
__global__ __launch_bounds__(64) void attn_kernel(
    const unsigned short* __restrict__ qws,
    const unsigned short* __restrict__ kws,
    const unsigned short* __restrict__ vws,
    const float* __restrict__ x,
    const float* __restrict__ gamma,
    float* __restrict__ out)
{
    extern __shared__ unsigned short smem_us[];

    const int tid  = threadIdx.x;
    const int lane = tid & 31;
    const int hi   = lane >> 4;
    const int col  = lane & 15;
    const int i0   = blockIdx.x * 16;
    const int b    = blockIdx.y;
#if USE_TDM
    const int w    = __builtin_amdgcn_readfirstlane(tid >> 5);
#else
    const int w    = tid >> 5;
#endif
    const int obase = w * 128;

    // Q as B-fragment: two 16B loads per lane (k 8h..8h+7, 16+8h..), reused all steps.
    const unsigned short* qrow = qws + ((size_t)b * NPIX + i0 + col) * CQKD;
    const v16bf qb = cat16(qrow + 8 * hi, qrow + 8 * hi + 16);

    float m_s = -3.0e38f, l_s = 0.0f;
    v8f acc[8];
    #pragma unroll
    for (int t = 0; t < 8; ++t) acc[t] = (v8f){};

    const unsigned short* kbase = kws + (size_t)b * NPIX * CQKD;
    const unsigned short* vbase = vws + (size_t)b * COD * NPIX;
    const int NSTEP = NPIX / 64;

#if USE_TDM
    const unsigned ldsbase = __builtin_amdgcn_groupstaticsize();
    const unsigned koff0 = ldsbase + (unsigned)(2 * w) * KBYTES;
    const unsigned short* ksm = smem_us + (size_t)(2 * w) * (KBYTES / 2);
    // Prologue: DMA K tile 0 (64 rows x 32 chans, padded rows in LDS).
    tdm_load_2d(koff0, kbase, 32, 64, 32, /*pad_int=*/3, /*pad_amt=*/3);
#endif

    #pragma unroll 1
    for (int step = 0; step < NSTEP; ++step) {
        const int j0 = step * 64;
#if USE_TDM
        if (step + 1 < NSTEP)
            tdm_load_2d(koff0 + (unsigned)((step + 1) & 1) * KBYTES,
                        kbase + (size_t)(j0 + 64) * CQKD, 32, 64, 32, 3, 3);
        // In-order per-wave TDM: <=2 outstanding ==> current tile landed.
        if (step + 1 < NSTEP) __builtin_amdgcn_s_wait_tensorcnt((short)2);
        else                  __builtin_amdgcn_s_wait_tensorcnt((short)0);
        const unsigned short* kt = ksm + (size_t)(step & 1) * (KBYTES / 2);
#endif
        // S^T tiles: rows j (VGPR axis), cols i (lanes); 4 tiles = 64 keys.
        v8f s[4];
        #pragma unroll
        for (int t = 0; t < 4; ++t) {
#if USE_TDM
            const unsigned short* kr = kt + (size_t)(t * 16 + col) * KPITCH + 8 * hi;
#else
            const unsigned short* kr =
                kbase + (size_t)(j0 + t * 16 + col) * CQKD + 8 * hi;
#endif
            const v16bf ka = cat16(kr, kr + 16);
            const v8f z = {};
            s[t] = __builtin_amdgcn_wmma_f32_16x16x32_bf16(false, ka, false, qb,
                                                           (short)0, z, false, false);
        }
#if !USE_TDM
        if (j0 + 64 < NPIX)
            __builtin_prefetch(kbase + (size_t)(j0 + 64 + col) * CQKD, 0, 0);
#endif

        // Online softmax over keys: serial in-register reduce + one half-swap.
        float rm = s[0][0];
        #pragma unroll
        for (int t = 0; t < 4; ++t)
            #pragma unroll
            for (int v = 0; v < 8; ++v) rm = fmaxf(rm, s[t][v]);
        rm = fmaxf(rm, __shfl_xor(rm, 16, 32));
        const float mnew = fmaxf(m_s, rm);
        const float sc = __expf(m_s - mnew);
        m_s = mnew;

        float rs = 0.0f;
        #pragma unroll
        for (int t = 0; t < 4; ++t)
            #pragma unroll
            for (int v = 0; v < 8; ++v) {
                const float p = __expf(s[t][v] - mnew);
                s[t][v] = p;
                rs += p;
            }
        rs += __shfl_xor(rs, 16, 32);
        l_s = l_s * sc + rs;

        #pragma unroll
        for (int t = 0; t < 8; ++t)
            #pragma unroll
            for (int v = 0; v < 8; ++v) acc[t][v] *= sc;

        // Pack P^T tiles to bf16x2 dwords.
        unsigned pk[4][4];
        #pragma unroll
        for (int t = 0; t < 4; ++t)
            #pragma unroll
            for (int u = 0; u < 4; ++u)
                pk[t][u] = pack_bf16(s[t][2 * u], s[t][2 * u + 1]);

        // Two O^T WMMA passes (32 keys each): B-frag via half-swap shuffles.
        #pragma unroll
        for (int f = 0; f < 2; ++f) {
            unsigned fr[8];
            #pragma unroll
            for (int u = 0; u < 4; ++u) {
                const unsigned snd = hi ? pk[2 * f][u] : pk[2 * f + 1][u];
                const unsigned rec = __shfl_xor(snd, 16, 32);
                fr[u]     = hi ? rec : pk[2 * f][u];
                fr[u + 4] = hi ? pk[2 * f + 1][u] : rec;
            }
            const v16bf pb = __builtin_bit_cast(
                v16bf, (uv8){fr[0], fr[1], fr[2], fr[3], fr[4], fr[5], fr[6], fr[7]});
            #pragma unroll
            for (int t = 0; t < 8; ++t) {
                const unsigned short* vp =
                    vbase + (size_t)(obase + t * 16 + col) * NPIX + j0 + 32 * f + 16 * hi;
                const v16bf av = cat16(vp, vp + 8);
                acc[t] = __builtin_amdgcn_wmma_f32_16x16x32_bf16(false, av, false, pb,
                                                                 (short)0, acc[t], false, false);
            }
        }
    }

    // Epilogue: out = (gamma/l)*acc + x ; lanes span pixels -> coalesced.
    const float gl = gamma[0] / l_s;
    #pragma unroll
    for (int t = 0; t < 8; ++t)
        #pragma unroll
        for (int v = 0; v < 8; ++v) {
            const int och  = obase + t * 16 + v + (hi << 3);
            const int ipix = i0 + col;
            const size_t idx = ((size_t)b * COD + och) * NPIX + ipix;
            out[idx] = gl * acc[t][v] + x[idx];
        }
}

extern "C" void kernel_launch(void* const* d_in, const int* in_sizes, int n_in,
                              void* d_out, int out_size, void* d_ws, size_t ws_size,
                              hipStream_t stream) {
    const float* x     = (const float*)d_in[0];
    const float* Wq    = (const float*)d_in[1];
    const float* bq    = (const float*)d_in[2];
    const float* Wk    = (const float*)d_in[3];
    const float* bk    = (const float*)d_in[4];
    const float* Wv    = (const float*)d_in[5];
    const float* bv    = (const float*)d_in[6];
    const float* gamma = (const float*)d_in[7];
    float* out = (float*)d_out;

    unsigned short* qws = (unsigned short*)d_ws;                 // 2 MB
    unsigned short* kws = qws + (size_t)BDIM * NPIX * CQKD;      // 2 MB
    unsigned short* vws = kws + (size_t)BDIM * NPIX * CQKD;      // 16 MB

    dim3 g1(NPIX / 16, (CQKD + CQKD + COD) / 16, BDIM);
    hipLaunchKernelGGL(proj_kernel, g1, dim3(32), 0, stream,
                       x, Wq, bq, Wk, bk, Wv, bv, qws, kws, vws);

    dim3 g2(NPIX / 16, BDIM);
    hipLaunchKernelGGL(attn_kernel, g2, dim3(64), ATTN_SMEM, stream,
                       qws, kws, vws, x, gamma, out);
}
